// SplitMonoto_25305947308125
// MI455X (gfx1250) — compile-verified, matrix operand order checked by reading, not run on hardware
//
#include <hip/hip_runtime.h>
#include <hip/hip_bf16.h>

// ---- problem constants (match reference) ----
#define Dd 16
#define Hh 256
#define Mm 3          // number of HxH hidden layers
#define NN 65536
#define PADH 264      // halves per LDS row (256 + 8 pad to stagger banks)
#define MT 4          // 16-row M-tiles per wave (64 samples)

typedef __attribute__((ext_vector_type(16))) _Float16 v16h;
typedef __attribute__((ext_vector_type(8)))  float    v8f;
typedef __attribute__((ext_vector_type(4)))  unsigned int u32x4;

union FragH { v16h h; u32x4 q[2]; };

// ---- tanh: prefer gfx1250 hardware v_tanh, fall back to exp/rcp ----
__device__ __forceinline__ float tanh_sw(float x) {
    x = fminf(fmaxf(x, -9.0f), 9.0f);
    float e = __expf(2.0f * x);
    return (e - 1.0f) * __builtin_amdgcn_rcpf(e + 1.0f);
}
#if __has_builtin(__builtin_amdgcn_tanhf)
__device__ __forceinline__ float tanh_fast(float x) { return __builtin_amdgcn_tanhf(x); }
#else
__device__ __forceinline__ float tanh_fast(float x) { return tanh_sw(x); }
#endif
#if __has_builtin(__builtin_amdgcn_tanh_f16)
__device__ __forceinline__ _Float16 tanh_h(float x) {
    return __builtin_amdgcn_tanh_f16((_Float16)x);   // v_cvt + v_tanh_f16
}
#else
__device__ __forceinline__ _Float16 tanh_h(float x) { return (_Float16)tanh_fast(x); }
#endif

// ---------------------------------------------------------------------------
// Kernel 1: pack W_mid (f32, [M][D][H][H], y[k] = sum_h h[h]*W[h][k]) into
// f16 B-fragment-major layout. Fragment (m,d,kt,nt) = 1024B = 32 lanes x 8
// dwords, matching the CDNA5 WMMA 16x16x32 f16 B-operand per-lane layout:
//   lane L holds column N = nt*16 + (L&15); lane group hi = L>>4
//   dword v<4 : K = kt*32 +      hi*8 + 2v + p
//   dword v>=4: K = kt*32 + 16 + hi*8 + 2(v-4) + p
// ---------------------------------------------------------------------------
__global__ __launch_bounds__(256) void pack_w_kernel(
    const float* __restrict__ W_mid, unsigned int* __restrict__ packed) {
    int tid  = blockIdx.x * blockDim.x + threadIdx.x;
    int v    = tid & 7;
    int lane = (tid >> 3) & 31;
    int nt   = (tid >> 8) & 15;
    int kt   = (tid >> 12) & 7;
    int d    = (tid >> 15) & 15;
    int m    = tid >> 19;
    if (m >= Mm) return;

    int Nl   = lane & 15;
    int hi   = (lane >> 4) & 1;
    int Ncol = nt * 16 + Nl;
    int K    = kt * 32 + hi * 8 + ((v & 4) ? 16 : 0) + 2 * (v & 3);

    const float* W = W_mid + (size_t)(m * Dd + d) * Hh * Hh;
    _Float16 h0 = (_Float16)W[(size_t)K * Hh + Ncol];
    _Float16 h1 = (_Float16)W[(size_t)(K + 1) * Hh + Ncol];
    unsigned u = ((unsigned)__builtin_bit_cast(unsigned short, h1) << 16) |
                 (unsigned)__builtin_bit_cast(unsigned short, h0);

    size_t frag = ((size_t)(m * Dd + d) * 8 + kt) * 16 + nt;   // fragment id
    packed[frag * 256 + lane * 8 + v] = u;                      // 256 dwords/frag
}

// ---------------------------------------------------------------------------
// Kernel 2: ymin = mlp(-1), ymax = mlp(+1) per dim, plain f32 (tiny work).
// yminmax layout: [0..15] = ymin (x=-1), [16..31] = ymax (x=+1)
// ---------------------------------------------------------------------------
__global__ __launch_bounds__(32) void minmax_kernel(
    const float* __restrict__ Win,  const float* __restrict__ bin,
    const float* __restrict__ Wmid, const float* __restrict__ bmid,
    const float* __restrict__ Wout, const float* __restrict__ bout,
    float* __restrict__ yminmax) {
    int d = blockIdx.x;
    int lane = threadIdx.x;            // single wave32
    __shared__ float hbuf[2][Hh];

    for (int e = 0; e < 2; ++e) {
        float x0 = (e == 0) ? -1.0f : 1.0f;
        for (int j = 0; j < 8; ++j) {
            int f = lane + 32 * j;
            hbuf[0][f] = tanh_fast(fmaf(x0, Win[d * Hh + f], bin[d * Hh + f]));
        }
        __syncthreads();
        int cur = 0;
        for (int m = 0; m < Mm; ++m) {
            float acc[8];
            for (int j = 0; j < 8; ++j)
                acc[j] = bmid[(m * Dd + d) * Hh + lane + 32 * j];
            const float* W = Wmid + (size_t)(m * Dd + d) * Hh * Hh;
            for (int hh = 0; hh < Hh; ++hh) {
                float hv = hbuf[cur][hh];
                const float* Wr = W + (size_t)hh * Hh;
                #pragma unroll
                for (int j = 0; j < 8; ++j)
                    acc[j] = fmaf(hv, Wr[lane + 32 * j], acc[j]);
            }
            for (int j = 0; j < 8; ++j)
                hbuf[cur ^ 1][lane + 32 * j] = tanh_fast(acc[j]);
            cur ^= 1;
            __syncthreads();
        }
        float s = 0.f;
        for (int j = 0; j < 8; ++j)
            s = fmaf(hbuf[cur][lane + 32 * j], Wout[d * Hh + lane + 32 * j], s);
        for (int off = 16; off >= 1; off >>= 1) s += __shfl_xor(s, off, 32);
        if (lane == 0) yminmax[e * Dd + d] = s + bout[d];
        __syncthreads();
    }
}

// ---------------------------------------------------------------------------
// Kernel 3: main MLP. grid = (N/64, D); block = 32 (one wave); each wave owns
// 64 samples = four 16-row M-tiles (4x B-fragment reuse -> ~6 GB total L2
// weight traffic). Hidden layers: nt blocked by 4 (A reused across 4 N-tiles),
// kt fully unrolled. Per layer per wave: 512 v_wmma_f32_16x16x32_f16,
// 256 ds_load_b128 (A), 256 global_load_b128 (B).
// ---------------------------------------------------------------------------
__global__ __launch_bounds__(32) void mlp_kernel(
    const float* __restrict__ x,
    const float* __restrict__ Win,  const float* __restrict__ bin,
    const float* __restrict__ bmid, const _Float16* __restrict__ packedW,
    const float* __restrict__ Wout, const float* __restrict__ bout,
    const float* __restrict__ yminmax, float* __restrict__ out) {
    const int d     = blockIdx.y;
    const int lane  = threadIdx.x & 31;
    const int Nl    = lane & 15;      // sample-in-tile / N-column-in-tile
    const int hi    = lane >> 4;      // lane group
    const int sbase = blockIdx.x * (16 * MT);

    // 2 ping-pong h buffers, 64 rows x 264 halves: 67,584 B (gfx1250: 320 KB max)
    __shared__ alignas(16) _Float16 smem[2][16 * MT * PADH];
    _Float16* buf0 = &smem[0][0];
    _Float16* buf1 = &smem[1][0];

    // ---- input layer: h = tanh(x * W_in + b_in) -> f16 -> LDS ----
    {
        const float* Wi = Win + d * Hh;
        const float* bi = bin + d * Hh;
        #pragma unroll
        for (int t = 0; t < MT; ++t) {
            float xv = x[(size_t)d * NN + sbase + t * 16 + Nl];
            _Float16* row = buf0 + (t * 16 + Nl) * PADH;
            #pragma unroll 4
            for (int f0 = 0; f0 < 128; f0 += 2) {
                int f = hi * 128 + f0;
                row[f]     = tanh_h(fmaf(xv, Wi[f],     bi[f]));
                row[f + 1] = tanh_h(fmaf(xv, Wi[f + 1], bi[f + 1]));
            }
        }
    }
    __syncthreads();

    // ---- hidden layers on WMMA ----
    _Float16* cur = buf0;
    _Float16* nxt = buf1;
    for (int m = 0; m < Mm; ++m) {
        const _Float16* Wp = packedW + (size_t)(m * Dd + d) * 8 * 16 * 512;
        const float*    bm = bmid + (m * Dd + d) * Hh;
        for (int ntb = 0; ntb < 4; ++ntb) {
            v8f acc[MT][4];
            #pragma unroll
            for (int j = 0; j < 4; ++j) {
                float bias = bm[(ntb * 4 + j) * 16 + Nl];
                v8f b8 = {bias, bias, bias, bias, bias, bias, bias, bias};
                #pragma unroll
                for (int t = 0; t < MT; ++t) acc[t][j] = b8;
            }
            #pragma unroll
            for (int kt = 0; kt < 8; ++kt) {
                // A fragments for all M-tiles: lane holds row M = t*16+Nl,
                // halves at K = kt*32+hi*8+{0..7} and kt*32+16+hi*8+{0..7}
                FragH a[MT];
                #pragma unroll
                for (int t = 0; t < MT; ++t) {
                    const _Float16* ar =
                        cur + (t * 16 + Nl) * PADH + kt * 32 + hi * 8;
                    a[t].q[0] = *(const u32x4*)(ar);
                    a[t].q[1] = *(const u32x4*)(ar + 16);
                }
                #pragma unroll
                for (int j = 0; j < 4; ++j) {
                    int nt = ntb * 4 + j;
                    const _Float16* bp =
                        Wp + ((size_t)(kt * 16 + nt)) * 512 + lane * 16;
                    FragH b;
                    b.q[0] = *(const u32x4*)(bp);
                    b.q[1] = *(const u32x4*)(bp + 8);
                    #pragma unroll
                    for (int t = 0; t < MT; ++t)
                        acc[t][j] = __builtin_amdgcn_wmma_f32_16x16x32_f16(
                            false, a[t].h, false, b.h, (short)0, acc[t][j],
                            false, false);
                }
            }
            // C layout: VGPR r -> row M = t*16 + r + hi*8, col N = nt*16 + Nl
            #pragma unroll
            for (int t = 0; t < MT; ++t) {
                #pragma unroll
                for (int j = 0; j < 4; ++j) {
                    int ncol = (ntb * 4 + j) * 16 + Nl;
                    #pragma unroll
                    for (int r = 0; r < 8; ++r) {
                        int row = t * 16 + r + hi * 8;
                        nxt[row * PADH + ncol] = tanh_h(acc[t][j][r]);
                    }
                }
            }
        }
        __syncthreads();
        _Float16* tmp = cur; cur = nxt; nxt = tmp;
    }

    // ---- output layer + min-max normalization ----
    {
        const float* Wo = Wout + d * Hh;
        float ymin = yminmax[d];
        float ymax = yminmax[Dd + d];
        float inv  = __builtin_amdgcn_rcpf(ymax - ymin);
        #pragma unroll
        for (int t = 0; t < MT; ++t) {
            const _Float16* row = cur + (t * 16 + Nl) * PADH + hi * 128;
            float s = 0.f;
            #pragma unroll 8
            for (int f = 0; f < 128; ++f)
                s = fmaf((float)row[f], Wo[hi * 128 + f], s);
            s += __shfl_xor(s, 16, 32);
            if (hi == 0) {
                float y = s + bout[d];
                out[(size_t)d * NN + sbase + t * 16 + Nl] =
                    (y - ymin) * inv * 2.0f - 1.0f;
            }
        }
    }
}

// ---------------------------------------------------------------------------
extern "C" void kernel_launch(void* const* d_in, const int* in_sizes, int n_in,
                              void* d_out, int out_size, void* d_ws, size_t ws_size,
                              hipStream_t stream) {
    const float* x    = (const float*)d_in[0];   // [D,N,1]
    const float* Win  = (const float*)d_in[1];   // [D,1,H]
    const float* bin  = (const float*)d_in[2];   // [D,H]
    const float* Wmid = (const float*)d_in[3];   // [M,D,H,H]
    const float* bmid = (const float*)d_in[4];   // [M,D,H]
    const float* Wout = (const float*)d_in[5];   // [D,H,1]
    const float* bout = (const float*)d_in[6];   // [D,1]
    float* out = (float*)d_out;

    float*        yminmax = (float*)d_ws;                       // 32 floats
    _Float16*     packedW = (_Float16*)((char*)d_ws + 128);     // ~6 MB f16
    unsigned int* packedU = (unsigned int*)packedW;

    // one-shot weight pack: Mm*Dd*8*16 fragments * 256 dwords = 1,572,864 thr
    pack_w_kernel<<<6144, 256, 0, stream>>>(Wmid, packedU);
    // per-dim ymin/ymax (mlp at x = -1, +1)
    minmax_kernel<<<Dd, 32, 0, stream>>>(Win, bin, Wmid, bmid, Wout, bout, yminmax);
    // main forward pass: one wave / 64 samples per block
    mlp_kernel<<<dim3(NN / (16 * MT), Dd), 32, 0, stream>>>(
        x, Win, bin, bmid, packedW, Wout, bout, yminmax, out);
}
